// QLSTM_65481071399785
// MI455X (gfx1250) — compile-verified, matrix operand order checked by reading, not run on hardware
//
#include <hip/hip_runtime.h>

// ---------------------------------------------------------------------------
// QLSTM on MI455X (gfx1250): persistent per-batch-slab recurrence with
// V_WMMA_F32_16X16X32_F16 tile GEMMs. Batch rows never interact, so each
// workgroup owns 16 rows for all 1024 timesteps; h lives in LDS as f16,
// c lives in registers. Weights are repacked once into WMMA fragment layout
// and end up fully register-resident (compiler hoists them via VGPR MSBs).
// ---------------------------------------------------------------------------

typedef __attribute__((ext_vector_type(16))) _Float16 v16h;
typedef __attribute__((ext_vector_type(8)))  _Float16 v8h;
typedef __attribute__((ext_vector_type(4)))  _Float16 v4h;
typedef __attribute__((ext_vector_type(8)))  float    v8f;
typedef __attribute__((ext_vector_type(4)))  float    v4f;

#define SEQ    1024
#define BATCH  512
#define IND    128
#define HID    128
#define KTOT   256          // IN_DIM + HID
#define NTOT   512          // 4 gates * HID
#define MROWS  16           // batch rows per block
#define NBLK   (BATCH / MROWS)   // 32 blocks
#define NTILES (NTOT / 16)  // 32
#define KTILES (KTOT / 32)  // 8

// ws layout: [0, 256KB) packed f16 weights (NTILES*KTILES frags * 32 lanes * 16 halfs)
//            [256KB, +2KB) fused bias[512] f32
#define WS_BIAS_OFF (NTILES * KTILES * 32 * 16 * sizeof(_Float16))   // 262144

__device__ __forceinline__ float fast_sigmoid(float x) {
    return 1.0f / (1.0f + __expf(-x));
}
__device__ __forceinline__ float fast_tanh(float x) {
    float e = __expf(2.0f * x);
    return (e - 1.0f) / (e + 1.0f);
}

// ---------------------------------------------------------------------------
// Kernel 1: pack W[256,128] x4 (row-major, f32) into per-lane WMMA-B frags and
// fuse biases. Fragment layout (matches CDNA5 16-bit operand striping):
//   frag f = nt*8 + kt  (nt: 0..31 global N-tile, gate = nt>>3; kt: 0..7)
//   lane l: n = nt*16 + (l&15) within gate block; kbase = kt*32 + (l>>4)*8
//   halfs j=0..7  -> K = kbase + j
//   halfs j=8..15 -> K = kbase + 16 + (j-8)
// ---------------------------------------------------------------------------
__global__ void qlstm_pack_kernel(const float* __restrict__ Wf,
                                  const float* __restrict__ Wi,
                                  const float* __restrict__ Wg,
                                  const float* __restrict__ Wo,
                                  const float* __restrict__ bf,
                                  const float* __restrict__ bi,
                                  const float* __restrict__ bg,
                                  const float* __restrict__ bo,
                                  _Float16* __restrict__ pk,
                                  float* __restrict__ biasAll) {
    if (blockIdx.x < 32) {
        int gt   = blockIdx.x * 256 + threadIdx.x;   // 8192 threads = 256 frags * 32 lanes
        int f    = gt >> 5;
        int lane = gt & 31;
        int nt   = f >> 3;            // 0..31
        int kt   = f & 7;             // 0..7
        int gate = nt >> 3;
        const float* W = (gate == 0) ? Wf : (gate == 1) ? Wi : (gate == 2) ? Wg : Wo;
        int n     = ((nt & 7) * 16) + (lane & 15);   // column within this gate's [256,128]
        int kbase = kt * 32 + ((lane >> 4) * 8);
        _Float16* dst = pk + ((size_t)f * 32 + lane) * 16;
        #pragma unroll
        for (int j = 0; j < 8; ++j)
            dst[j] = (_Float16)W[(size_t)(kbase + j) * 128 + n];
        #pragma unroll
        for (int j = 0; j < 8; ++j)
            dst[8 + j] = (_Float16)W[(size_t)(kbase + 16 + j) * 128 + n];
    } else {
        // fused bias: [f | i | g | o] -> biasAll[512]
        for (int i = threadIdx.x; i < NTOT; i += 256) {
            int g = i >> 7, j = i & 127;
            const float* b = (g == 0) ? bf : (g == 1) ? bi : (g == 2) ? bg : bo;
            biasAll[i] = b[j];
        }
    }
}

// ---------------------------------------------------------------------------
// Kernel 2: persistent recurrence. 32 blocks x 256 threads (8 waves).
// Block owns rows [blockIdx*16, +16). Per step:
//   GEMM: M=16, N=512, K=256 -> 1 mtile x 32 ntiles x 8 ktiles = 256 WMMAs,
//   split 4 ntiles/wave, A fragments double-buffered through LDS.
//   Then cumprod(cos(f_lin)) scan + LSTM update.
// ---------------------------------------------------------------------------
__global__ void __launch_bounds__(256)
qlstm_scan_kernel(const float* __restrict__ xin,      // [SEQ, BATCH, IND]
                  const _Float16* __restrict__ pk,    // packed weights
                  const float* __restrict__ biasAll,  // [512]
                  float* __restrict__ out) {          // outputs | hx | cx
    __shared__ _Float16 s_comb[MROWS][KTOT + 16];     // [16][272] f16: x | h
    __shared__ float    s_gates[MROWS][NTOT + 4];     // [16][516] f32: f|i|g|o

    const int tid  = threadIdx.x;
    const int wave = tid >> 5;
    const int lane = tid & 31;
    const int r0   = blockIdx.x * MROWS;

    const size_t OUT_HX = (size_t)SEQ * BATCH * HID;
    const size_t OUT_CX = OUT_HX + (size_t)BATCH * HID;

    // Per-lane persistent cell state: wave handles rows 2w, 2w+1; lane covers
    // hidden h = lane*4 .. lane*4+3 of each.
    float c_reg[2][4];
    #pragma unroll
    for (int rr = 0; rr < 2; ++rr)
        #pragma unroll
        for (int j = 0; j < 4; ++j) c_reg[rr][j] = 0.0f;

    // Preload this lane's 4 bias values (one per owned N-tile; column-constant).
    float bia[4];
    #pragma unroll
    for (int j = 0; j < 4; ++j)
        bia[j] = biasAll[(wave * 4 + j) * 16 + (lane & 15)];

    // Zero h half of comb (h_0 = 0).
    for (int i = tid; i < MROWS * HID; i += 256)
        s_comb[i >> 7][IND + (i & 127)] = (_Float16)0.0f;

    const v16h* Bp = (const v16h*)pk;

    for (int t = 0; t < SEQ; ++t) {
        // ---- stage x_t slab into LDS as f16 (2048 floats, 8/thread) ----
        {
            const float* xp = xin + ((size_t)t * BATCH + r0) * IND + tid * 8;
            int r = (tid * 8) >> 7, c = (tid * 8) & 127;
            v4f a = *(const v4f*)xp;
            v4f b = *(const v4f*)(xp + 4);
            v8h hv;
            #pragma unroll
            for (int j = 0; j < 4; ++j) { hv[j] = (_Float16)a[j]; hv[4 + j] = (_Float16)b[j]; }
            *(v8h*)&s_comb[r][c] = hv;
            if (t + 1 < SEQ)
                __builtin_prefetch(xp + (size_t)BATCH * IND, 0, 3);   // WGP-scope prefetch
        }
        __syncthreads();   // comb (x + h from prev step) ready; gates free (WAR)

        // ---- WMMA GEMM: this wave's 4 N-tiles over 8 K-tiles ----
        v8f acc[4];
        #pragma unroll
        for (int j = 0; j < 4; ++j) acc[j] = (v8f)0.0f;

        const int m    = lane & 15;
        const int ksub = (lane >> 4) * 8;

        // A fragments double-buffered: issue loads for kt+1 before WMMAs of kt
        union AF { v16h v; v8h h8[2]; };
        AF Abuf[2];
        Abuf[0].h8[0] = *(const v8h*)&s_comb[m][ksub];
        Abuf[0].h8[1] = *(const v8h*)&s_comb[m][ksub + 16];
        #pragma unroll
        for (int kt = 0; kt < KTILES; ++kt) {
            if (kt + 1 < KTILES) {
                const int kb = (kt + 1) * 32 + ksub;
                Abuf[(kt + 1) & 1].h8[0] = *(const v8h*)&s_comb[m][kb];
                Abuf[(kt + 1) & 1].h8[1] = *(const v8h*)&s_comb[m][kb + 16];
            }
            const v16h Av = Abuf[kt & 1].v;
            #pragma unroll
            for (int j = 0; j < 4; ++j) {
                const int nt = wave * 4 + j;
                v16h B = Bp[(size_t)(nt * KTILES + kt) * 32 + lane];
                acc[j] = __builtin_amdgcn_wmma_f32_16x16x32_f16(
                    false, Av, false, B, (short)0, acc[j], false, false);
            }
        }

        // ---- scatter C + bias into LDS gate buffer ----
        #pragma unroll
        for (int j = 0; j < 4; ++j) {
            const int n = (wave * 4 + j) * 16 + (lane & 15);
            #pragma unroll
            for (int v = 0; v < 8; ++v) {
                const int mm = ((lane >> 4) * 8) + v;
                s_gates[mm][n] = acc[j][v] + bia[j];
            }
        }
        __syncthreads();   // gates complete; comb-x safe to overwrite next iter

        // ---- elementwise + cumprod scan: wave handles rows 2w, 2w+1 ----
        #pragma unroll
        for (int rr = 0; rr < 2; ++rr) {
            const int r_loc = wave * 2 + rr;
            const int h0    = lane * 4;
            const float* gr = &s_gates[r_loc][0];
            v4f fl = *(const v4f*)&gr[h0];
            v4f il = *(const v4f*)&gr[HID + h0];
            v4f gl = *(const v4f*)&gr[2 * HID + h0];
            v4f ol = *(const v4f*)&gr[3 * HID + h0];

            // in-lane cumulative product of cos(f_lin)
            float p[4];
            p[0] = __cosf(fl[0]);
            p[1] = p[0] * __cosf(fl[1]);
            p[2] = p[1] * __cosf(fl[2]);
            p[3] = p[2] * __cosf(fl[3]);

            // wave32 inclusive prefix product of per-lane totals
            float incl = p[3];
            #pragma unroll
            for (int d = 1; d < 32; d <<= 1) {
                float vsh = __shfl_up(incl, d, 32);
                if (lane >= d) incl *= vsh;
            }
            float excl = __shfl_up(incl, 1, 32);
            if (lane == 0) excl = 1.0f;

            v4f hv4, cv4;
            #pragma unroll
            for (int j = 0; j < 4; ++j) {
                float f = (excl * p[j] + 1.0f) * 0.5f;
                float i = (__cosf(il[j]) + 1.0f) * 0.5f;
                float g = fast_tanh(gl[j]);
                float o = fast_sigmoid(ol[j]);
                float c = f * c_reg[rr][j] + i * g;
                c_reg[rr][j] = c;
                cv4[j] = c;
                hv4[j] = o * fast_tanh(c);
            }

            // h_t -> output stream
            const int rg = r0 + r_loc;
            *(v4f*)&out[((size_t)t * BATCH + rg) * HID + h0] = hv4;

            // h_t (f16) back into comb for next step's GEMM
            v4h hh;
            #pragma unroll
            for (int j = 0; j < 4; ++j) hh[j] = (_Float16)hv4[j];
            *(v4h*)&s_comb[r_loc][IND + h0] = hh;

            if (t == SEQ - 1) {
                *(v4f*)&out[OUT_HX + (size_t)rg * HID + h0] = hv4;
                *(v4f*)&out[OUT_CX + (size_t)rg * HID + h0] = cv4;
            }
        }
        // next iteration's __syncthreads() (after x staging) orders
        // s_comb h-writes vs. GEMM reads and s_gates reads vs. writes.
    }
}

extern "C" void kernel_launch(void* const* d_in, const int* in_sizes, int n_in,
                              void* d_out, int out_size, void* d_ws, size_t ws_size,
                              hipStream_t stream) {
    (void)in_sizes; (void)n_in; (void)out_size; (void)ws_size;
    const float* xin = (const float*)d_in[0];
    const float* Wf  = (const float*)d_in[1];
    const float* bf  = (const float*)d_in[2];
    const float* Wi  = (const float*)d_in[3];
    const float* bi  = (const float*)d_in[4];
    const float* Wg  = (const float*)d_in[5];
    const float* bg  = (const float*)d_in[6];
    const float* Wo  = (const float*)d_in[7];
    const float* bo  = (const float*)d_in[8];

    _Float16* pk      = (_Float16*)d_ws;
    float*    biasAll = (float*)((char*)d_ws + WS_BIAS_OFF);
    float*    out     = (float*)d_out;

    qlstm_pack_kernel<<<33, 256, 0, stream>>>(Wf, Wi, Wg, Wo, bf, bi, bg, bo, pk, biasAll);
    qlstm_scan_kernel<<<NBLK, 256, 0, stream>>>(xin, pk, biasAll, out);
}
